// AttentionBlock_52450140619304
// MI455X (gfx1250) — matrix operand order repaired and checked by reading, not compiled, not run on hardware
//
#include <hip/hip_runtime.h>
#include <hip/hip_bf16.h>

typedef __attribute__((ext_vector_type(16))) _Float16 v16h;
typedef __attribute__((ext_vector_type(8)))  _Float16 v8h;
typedef __attribute__((ext_vector_type(4)))  _Float16 v4h;
typedef __attribute__((ext_vector_type(8)))  float    v8f;

#define B_  16
#define C_  512
#define S_  1024   // H*W
#define OC_ 1536   // 3*C

// ---------------------------------------------------------------------------
// helpers
// ---------------------------------------------------------------------------
static __device__ inline v16h combine16(v8h lo, v8h hi) {
  return __builtin_shufflevector(lo, hi, 0,1,2,3,4,5,6,7,8,9,10,11,12,13,14,15);
}

// A fragment (16x32 f16, row in LDS, contiguous K):
// lane<16 holds K {0..7, 16..23}; lane>=16 holds K {8..15, 24..31}
static __device__ inline v16h load_fragA(const _Float16* row, int lane) {
  const _Float16* p = row + ((lane & 16) ? 8 : 0);
  return combine16(*(const v8h*)p, *(const v8h*)(p + 16));
}
// B fragment (32x16 f16 from Bt row = column of B, contiguous K):
// lane<16 holds K {0..7, 16..23}; lane>=16 holds K {8..15, 24..31}
static __device__ inline v16h load_fragB(const _Float16* row, int lane) {
  const _Float16* p = row + ((lane & 16) ? 8 : 0);
  return combine16(*(const v8h*)p, *(const v8h*)(p + 16));
}

// ---------------------------------------------------------------------------
// f32 -> f16 conversion (weights)
// ---------------------------------------------------------------------------
__global__ void cvt_f16(const float* __restrict__ s, _Float16* __restrict__ d, int n) {
  for (int i = blockIdx.x * blockDim.x + threadIdx.x; i < n; i += gridDim.x * blockDim.x)
    d[i] = (_Float16)s[i];
}

// ---------------------------------------------------------------------------
// GroupNorm: per (batch, group of 16 channels); writes xnT[b][s][c] as f16
// ---------------------------------------------------------------------------
__global__ __launch_bounds__(256)
void groupnorm_k(const float* __restrict__ x, const float* __restrict__ gw,
                 const float* __restrict__ gb, _Float16* __restrict__ xnT) {
  const int bg = blockIdx.x;
  const int b = bg >> 5, g = bg & 31;
  const int c0 = g * 16;
  const float* xp = x + ((long long)b * C_ + c0) * S_;   // 16 x 1024 contiguous

  __shared__ _Float16 tile[16 * S_];
  __shared__ float rs[8], rs2[8];

  float s = 0.f, s2 = 0.f;
  for (int i = threadIdx.x; i < 16 * S_; i += 256) {
    float v = xp[i]; s += v; s2 += v * v;
  }
  #pragma unroll
  for (int o = 16; o; o >>= 1) { s += __shfl_xor(s, o, 32); s2 += __shfl_xor(s2, o, 32); }
  const int wv = threadIdx.x >> 5, ln = threadIdx.x & 31;
  if (ln == 0) { rs[wv] = s; rs2[wv] = s2; }
  __syncthreads();
  s = 0.f; s2 = 0.f;
  #pragma unroll
  for (int i = 0; i < 8; ++i) { s += rs[i]; s2 += rs2[i]; }
  const float mean = s * (1.f / 16384.f);
  const float var  = s2 * (1.f / 16384.f) - mean * mean;
  const float inv  = rsqrtf(var + 1e-5f);

  for (int i = threadIdx.x; i < 16 * S_; i += 256) {
    int c = i >> 10;
    float sc = gw[c0 + c] * inv;
    float sh = gb[c0 + c] - mean * sc;
    tile[i] = (_Float16)(xp[i] * sc + sh);
  }
  __syncthreads();

  _Float16* op = xnT + (long long)b * S_ * C_ + c0;
  for (int sp = threadIdx.x; sp < S_; sp += 256) {
    v8h v0, v1;
    #pragma unroll
    for (int c = 0; c < 8; ++c) { v0[c] = tile[c * S_ + sp]; v1[c] = tile[(c + 8) * S_ + sp]; }
    *(v8h*)&op[(long long)sp * C_]     = v0;
    *(v8h*)&op[(long long)sp * C_ + 8] = v1;
  }
}

// ---------------------------------------------------------------------------
// Generic WMMA GEMM: D[M][N] = A[M][K] * Bt[N][K]^T  (A, Bt f16, K-contiguous)
// block tile 128x128, 8 waves, wave tile 64x32 (eight 16x16x32 WMMAs / K-step)
// Double-buffered LDS software pipeline: next tile's global loads issue before
// the current tile's WMMAs, ds_store into alternate buffer after compute.
// MODE 0: store f16 transposed  Dt[n][m]      (+bias by m)
// MODE 1: store f16 direct      D[m][n]       (+bias by m)
// MODE 2: store f32 * scale     D[m][n]
// MODE 3: store f32 + bias + residual  D[m][n]
// ---------------------------------------------------------------------------
template<int MODE, bool HAS_BIAS>
__global__ __launch_bounds__(256)
void gemm_wmma(const _Float16* __restrict__ A, int ldA, long long sA,
               const _Float16* __restrict__ Bt, int ldBt, long long sBt,
               void* __restrict__ D, int ldD, long long sD,
               const float* __restrict__ bias, float scale,
               const float* __restrict__ res, int K) {
  constexpr int BM = 128, BN = 128, BK = 32, LDSW = 40; // 80B row: 16B aligned, conflict-free
  constexpr int ABUF = BM * LDSW, BBUF = BN * LDSW;
  constexpr int BUFSZ = ABUF + BBUF;
  __shared__ _Float16 smem[2 * BUFSZ];                  // 40 KB

  const int z = blockIdx.z;
  A  += (long long)z * sA;
  Bt += (long long)z * sBt;

  const int tid  = threadIdx.x;
  const int lane = tid & 31, wave = tid >> 5;
  const int wm = wave >> 2, wn = wave & 3;       // 2 x 4 wave grid
  const int blockM = blockIdx.y * BM, blockN = blockIdx.x * BN;

  // per-thread staging coordinates: 64B per thread per K-step
  const int lrow = tid >> 2;            // 0..63
  const int lcol = (tid & 3) * 8;       // 0,8,16,24
  const int soff = lrow * LDSW + lcol;
  const _Float16* gA0 = A  + (long long)(blockM + lrow) * ldA  + lcol;
  const _Float16* gA1 = gA0 + 64ll * ldA;
  const _Float16* gB0 = Bt + (long long)(blockN + lrow) * ldBt + lcol;
  const _Float16* gB1 = gB0 + 64ll * ldBt;

  // preload tile 0 into buffer 0
  v8h ra0 = *(const v8h*)gA0, ra1 = *(const v8h*)gA1;
  v8h rb0 = *(const v8h*)gB0, rb1 = *(const v8h*)gB1;
  *(v8h*)&smem[soff]                    = ra0;
  *(v8h*)&smem[soff + 64 * LDSW]        = ra1;
  *(v8h*)&smem[ABUF + soff]             = rb0;
  *(v8h*)&smem[ABUF + soff + 64 * LDSW] = rb1;
  __syncthreads();

  const int NK = K / BK;
  v8f acc[4][2] = {};

  for (int kt = 0; kt < NK; ++kt) {
    // issue next tile's global loads (branch-free: step 0 on last iteration)
    const int adv = (kt + 1 < NK) ? BK : 0;
    gA0 += adv; gA1 += adv; gB0 += adv; gB1 += adv;
    ra0 = *(const v8h*)gA0; ra1 = *(const v8h*)gA1;
    rb0 = *(const v8h*)gB0; rb1 = *(const v8h*)gB1;
    __builtin_prefetch(gA0 + BK, 0, 0);
    __builtin_prefetch(gA1 + BK, 0, 0);
    __builtin_prefetch(gB0 + BK, 0, 0);
    __builtin_prefetch(gB1 + BK, 0, 0);

    // compute from current buffer while loads are in flight
    const int cofs = (kt & 1) ? BUFSZ : 0;
    const _Float16* cA = smem + cofs;
    const _Float16* cB = smem + cofs + ABUF;
    v16h af[4], bf[2];
    #pragma unroll
    for (int i = 0; i < 4; ++i)
      af[i] = load_fragA(&cA[(wm * 64 + i * 16 + (lane & 15)) * LDSW], lane);
    #pragma unroll
    for (int j = 0; j < 2; ++j)
      bf[j] = load_fragB(&cB[(wn * 32 + j * 16 + (lane & 15)) * LDSW], lane);

    #pragma unroll
    for (int i = 0; i < 4; ++i)
      #pragma unroll
      for (int j = 0; j < 2; ++j)
        acc[i][j] = __builtin_amdgcn_wmma_f32_16x16x32_f16(
            false, af[i], false, bf[j], (short)0, acc[i][j], false, false);

    // everyone finished reading buffer (kt+1)&1 one iteration ago
    __syncthreads();
    const int nofs = BUFSZ - cofs;
    _Float16* nA = smem + nofs;
    _Float16* nB = smem + nofs + ABUF;
    *(v8h*)&nA[soff]             = ra0;
    *(v8h*)&nA[soff + 64 * LDSW] = ra1;
    *(v8h*)&nB[soff]             = rb0;
    *(v8h*)&nB[soff + 64 * LDSW] = rb1;
    __syncthreads();
  }

  // epilogue. D 16x16 f32 layout: lane<16 -> M=r, lane>=16 -> M=8+r; N=lane%16
  const int n_l = lane & 15;
  const int mo  = (lane & 16) ? 8 : 0;
  #pragma unroll
  for (int i = 0; i < 4; ++i) {
    #pragma unroll
    for (int j = 0; j < 2; ++j) {
      const int m0 = blockM + wm * 64 + i * 16 + mo;
      const int n  = blockN + wn * 32 + j * 16 + n_l;
      v8f vv = acc[i][j];
      if (HAS_BIAS) {
        #pragma unroll
        for (int r = 0; r < 8; ++r) vv[r] += bias[m0 + r];
      }
      if (MODE == 0) {
        _Float16* Dp = (_Float16*)D + (long long)z * sD;
        v8h h;
        #pragma unroll
        for (int r = 0; r < 8; ++r) h[r] = (_Float16)vv[r];
        *(v8h*)&Dp[(long long)n * ldD + m0] = h;
      } else if (MODE == 1) {
        _Float16* Dp = (_Float16*)D + (long long)z * sD;
        #pragma unroll
        for (int r = 0; r < 8; ++r) Dp[(long long)(m0 + r) * ldD + n] = (_Float16)vv[r];
      } else if (MODE == 2) {
        float* Dp = (float*)D + (long long)z * sD;
        #pragma unroll
        for (int r = 0; r < 8; ++r) Dp[(long long)(m0 + r) * ldD + n] = vv[r] * scale;
      } else {
        float* Dp = (float*)D + (long long)z * sD;
        const float* Rp = res + (long long)z * sD;
        #pragma unroll
        for (int r = 0; r < 8; ++r) {
          long long idx = (long long)(m0 + r) * ldD + n;
          Dp[idx] = vv[r] + Rp[idx];
        }
      }
    }
  }
}

// ---------------------------------------------------------------------------
// Row softmax over f32 scores; writes f16 probs in place (first half of row)
// ---------------------------------------------------------------------------
static __device__ inline float block_reduce(float v, bool is_max, float* red) {
  #pragma unroll
  for (int o = 16; o; o >>= 1) {
    float t = __shfl_xor(v, o, 32);
    v = is_max ? fmaxf(v, t) : v + t;
  }
  const int wv = threadIdx.x >> 5, ln = threadIdx.x & 31;
  __syncthreads();
  if (ln == 0) red[wv] = v;
  __syncthreads();
  v = red[0];
  #pragma unroll
  for (int i = 1; i < 8; ++i) v = is_max ? fmaxf(v, red[i]) : v + red[i];
  return v;
}

__global__ __launch_bounds__(256)
void softmax_rows(float* __restrict__ S) {
  __shared__ float red[8];
  const long long row = blockIdx.x;
  float* sp = S + row * S_;
  float4 v = ((const float4*)sp)[threadIdx.x];        // 256 threads * 4 = 1024
  float m = fmaxf(fmaxf(v.x, v.y), fmaxf(v.z, v.w));
  m = block_reduce(m, true, red);
  float e0 = __expf(v.x - m), e1 = __expf(v.y - m);
  float e2 = __expf(v.z - m), e3 = __expf(v.w - m);
  float s = block_reduce(e0 + e1 + e2 + e3, false, red);
  const float inv = 1.f / s;
  __syncthreads();                                    // all reads done before f16 overwrite
  _Float16* pp = (_Float16*)sp;
  v4h h = { (_Float16)(e0 * inv), (_Float16)(e1 * inv),
            (_Float16)(e2 * inv), (_Float16)(e3 * inv) };
  *(v4h*)&pp[threadIdx.x * 4] = h;
}

// ---------------------------------------------------------------------------
// launcher
// ---------------------------------------------------------------------------
extern "C" void kernel_launch(void* const* d_in, const int* in_sizes, int n_in,
                              void* d_out, int out_size, void* d_ws, size_t ws_size,
                              hipStream_t stream) {
  (void)in_sizes; (void)n_in; (void)out_size; (void)ws_size;
  const float* x      = (const float*)d_in[0];
  const float* norm_w = (const float*)d_in[1];
  const float* norm_b = (const float*)d_in[2];
  const float* qkv_w  = (const float*)d_in[3];
  const float* qkv_b  = (const float*)d_in[4];
  const float* proj_w = (const float*)d_in[5];
  const float* proj_b = (const float*)d_in[6];
  float* out = (float*)d_out;

  char* ws = (char*)d_ws;
  size_t off = 0;
  auto take = [&](size_t bytes) -> char* {
    char* p = ws + off; off = (off + bytes + 255) & ~(size_t)255; return p;
  };
  _Float16* wqkv16 = (_Float16*)take((size_t)OC_ * C_ * 2);
  _Float16* wproj16= (_Float16*)take((size_t)C_ * C_ * 2);
  _Float16* xnT    = (_Float16*)take((size_t)B_ * S_ * C_ * 2);   // [b][s][c]
  _Float16* G      = (_Float16*)take((size_t)B_ * S_ * 1024 * 2); // [b][s][q(512)|k(512)]
  _Float16* V      = (_Float16*)take((size_t)B_ * C_ * S_ * 2);   // [b][cv][s]
  float*    Sc     = (float*)   take((size_t)B_ * S_ * S_ * 4);   // scores; P aliased f16
  _Float16* Ht     = (_Float16*)take((size_t)B_ * S_ * C_ * 2);   // [b][s][c]

  // weights -> f16
  cvt_f16<<<512, 256, 0, stream>>>(qkv_w,  wqkv16,  OC_ * C_);
  cvt_f16<<<512, 256, 0, stream>>>(proj_w, wproj16, C_ * C_);

  // groupnorm -> xnT
  groupnorm_k<<<B_ * 32, 256, 0, stream>>>(x, norm_w, norm_b, xnT);

  // q,k GEMM (rows 0..1023 of Wqkv): D[o][s] stored transposed into G[b][s][o]
  gemm_wmma<0, true><<<dim3(8, 8, B_), 256, 0, stream>>>(
      wqkv16, C_, 0, xnT, C_, (long long)S_ * C_,
      (void*)G, 1024, (long long)S_ * 1024, qkv_b, 1.f, nullptr, C_);

  // v GEMM (rows 1024..1535): stored direct into V[b][cv][s]
  gemm_wmma<1, true><<<dim3(8, 4, B_), 256, 0, stream>>>(
      wqkv16 + 1024 * C_, C_, 0, xnT, C_, (long long)S_ * C_,
      (void*)V, S_, (long long)C_ * S_, qkv_b + 1024, 1.f, nullptr, C_);

  // scores: A = q rows of G, Bt = k rows of G; f32 * (1/sqrt(C))
  gemm_wmma<2, false><<<dim3(8, 8, B_), 256, 0, stream>>>(
      G, 1024, (long long)S_ * 1024, G + 512, 1024, (long long)S_ * 1024,
      (void*)Sc, S_, (long long)S_ * S_, nullptr, 0.044194173824159216f, nullptr, C_);

  // softmax rows, f16 P written in place (row pitch stays 4096B -> ld 2048 halves)
  softmax_rows<<<B_ * S_, 256, 0, stream>>>(Sc);

  // h GEMM: D[cv][n] = V[cv][:] * P[n][:]^T, stored transposed -> Ht[b][n][cv]
  gemm_wmma<0, false><<<dim3(8, 4, B_), 256, 0, stream>>>(
      V, S_, (long long)C_ * S_, (const _Float16*)Sc, 2048, (long long)S_ * 2048,
      (void*)Ht, C_, (long long)S_ * C_, nullptr, 1.f, nullptr, S_);

  // proj GEMM + bias + residual -> out[b][o][s] f32
  gemm_wmma<3, true><<<dim3(8, 4, B_), 256, 0, stream>>>(
      wproj16, C_, 0, Ht, C_, (long long)S_ * C_,
      (void*)out, S_, (long long)C_ * S_, proj_b, 1.f, x, C_);
}
